// AttentionBlock_42666205119071
// MI455X (gfx1250) — matrix operand order repaired and checked by reading, not compiled, not run on hardware
//
#include <hip/hip_runtime.h>
#include <hip/hip_bf16.h>

typedef __bf16 bf16_t;
typedef __attribute__((ext_vector_type(16))) __bf16 v16bf;
typedef __attribute__((ext_vector_type(8)))  __bf16 v8bf;
typedef __attribute__((ext_vector_type(4)))  __bf16 v4bf;
typedef __attribute__((ext_vector_type(8)))  float  v8f;

#define TILE_K     64            // two WMMA K-steps per staged tile
#define BLK_M      128
#define BLK_N      128
#define LDS_K      72            // bf16 elems per LDS row (144B: 16B-aligned, conflict-free)
#define LDS_BUF    (128 * LDS_K)

// ---------------------------------------------------------------------------
// 128x128 bf16 GEMM core, BOTH operands K-major:
//   C(f32) += A[M,K] (row-major) * BT[N,K] (row-major, i.e. B transposed)
// 256 threads = 8 waves (2M x 4N); each wave: 64x32 via 4x2 WMMA accs.
// Double-buffered LDS, ONE barrier per 64-deep K-tile (16 WMMAs/barrier);
// all LDS traffic is b128.
// ---------------------------------------------------------------------------
__device__ __forceinline__ void gemm_core_128x128(
    const bf16_t* __restrict__ A,  int lda,
    const bf16_t* __restrict__ BT, int ldbt,
    int K, int m0, int n0,
    bf16_t* As, bf16_t* Bs, v8f (&acc)[4][2])
{
    const int tid  = threadIdx.x;
    const int lane = tid & 31;
    const int wave = tid >> 5;
    const int wm   = wave >> 2;          // 0..1 (M)
    const int wn   = wave & 3;           // 0..3 (N)
    const int la   = lane & 15;
    const int khA  = (lane >> 4) * 16;   // A frag K-half base: 0 or 16
    const int khB  = (lane >> 4) * 16;   // B frag K-half base: 0 or 16

    // staging map: 128x64 tile = 1024 chunks of 8 bf16; thread owns 4 chunks
    // chunk c = tid + j*256: row = (c>>3) = tid>>3 + 32*j, kc = (c&7)*8
    const int srow = tid >> 3;           // 0..31
    const int skc  = (tid & 7) * 8;      // 0..56
    const bf16_t* gA = A  + (size_t)(m0 + srow) * lda  + skc;
    const bf16_t* gB = BT + (size_t)(n0 + srow) * ldbt + skc;

    v8bf ra[4], rb[4];
    auto load_regs = [&](int k0) {
        #pragma unroll
        for (int j = 0; j < 4; ++j) {
            ra[j] = *(const v8bf*)(gA + (size_t)(32 * j) * lda  + k0);
            rb[j] = *(const v8bf*)(gB + (size_t)(32 * j) * ldbt + k0);
        }
    };

    load_regs(0);
    int buf = 0;
    for (int k0 = 0; k0 < K; k0 += TILE_K) {
        bf16_t* as = As + buf * LDS_BUF;
        bf16_t* bs = Bs + buf * LDS_BUF;
        #pragma unroll
        for (int j = 0; j < 4; ++j) {
            *(v8bf*)&as[(srow + 32 * j) * LDS_K + skc] = ra[j];
            *(v8bf*)&bs[(srow + 32 * j) * LDS_K + skc] = rb[j];
        }
        __syncthreads();

        if (k0 + TILE_K < K) {
            load_regs(k0 + TILE_K);                    // overlaps WMMA compute
            if (k0 + 2 * TILE_K < K) {                 // prefetch 2 tiles ahead
                __builtin_prefetch(gA + k0 + 2 * TILE_K, 0, 3);
                __builtin_prefetch(gB + k0 + 2 * TILE_K, 0, 3);
            }
        }

        // two WMMA K-steps over the staged 64-deep tile
        #pragma unroll
        for (int ks = 0; ks < TILE_K; ks += 32) {
            v16bf afrag[4];
            #pragma unroll
            for (int mi = 0; mi < 4; ++mi) {
                int row = wm * 64 + mi * 16 + la;
                v8bf lo = *(const v8bf*)&as[row * LDS_K + ks + khA];        // K ..+7
                v8bf hi = *(const v8bf*)&as[row * LDS_K + ks + khA + 8];    // K +8..+15
                v16bf a;
                #pragma unroll
                for (int i = 0; i < 8; ++i) { a[i] = lo[i]; a[i + 8] = hi[i]; }
                afrag[mi] = a;
            }
            v16bf bfrag[2];
            #pragma unroll
            for (int ni = 0; ni < 2; ++ni) {
                int col = wn * 32 + ni * 16 + la;
                v8bf lo = *(const v8bf*)&bs[col * LDS_K + ks + khB];        // K ..+7
                v8bf hi = *(const v8bf*)&bs[col * LDS_K + ks + khB + 8];    // K +8..+15
                v16bf b;
                #pragma unroll
                for (int i = 0; i < 8; ++i) { b[i] = lo[i]; b[i + 8] = hi[i]; }
                bfrag[ni] = b;
            }
            #pragma unroll
            for (int mi = 0; mi < 4; ++mi)
                #pragma unroll
                for (int ni = 0; ni < 2; ++ni)
                    acc[mi][ni] = __builtin_amdgcn_wmma_f32_16x16x32_bf16(
                        false, afrag[mi], false, bfrag[ni],
                        (short)0, acc[mi][ni], false, false);
        }
        buf ^= 1;
    }
}
// accumulator coords: col = n0+wn*32+ni*16+(lane&15); rows = m0+wm*64+mi*16+(lane>>4)*8 + r

// ---------------------------------------------------------------------------
// K0a: vectorized f32 -> bf16 convert (weights; row-major A operands)
// ---------------------------------------------------------------------------
__global__ __launch_bounds__(256)
void f32_to_bf16_kernel(const float* __restrict__ in, bf16_t* __restrict__ out,
                        long long n4)
{
    long long i = (long long)blockIdx.x * 256 + threadIdx.x;
    if (i < n4) {
        float4 v = ((const float4*)in)[i];
        v4bf o;
        o[0] = (bf16_t)v.x; o[1] = (bf16_t)v.y;
        o[2] = (bf16_t)v.z; o[3] = (bf16_t)v.w;
        ((v4bf*)out)[i] = o;
    }
}

// ---------------------------------------------------------------------------
// K0b: tiled transpose-convert: x[b][c][s] f32 -> xT[b][s][c] bf16 (64x64 tiles)
// ---------------------------------------------------------------------------
__global__ __launch_bounds__(256)
void transpose_convert_kernel(const float* __restrict__ in, bf16_t* __restrict__ out)
{
    __shared__ __attribute__((aligned(16))) bf16_t tile[64 * 72];
    const int b  = blockIdx.z;
    const int c0 = blockIdx.y * 64;
    const int s0 = blockIdx.x * 64;
    const float* src = in + ((size_t)b * 1024 + c0) * 1024 + s0;
    const int tr  = threadIdx.x >> 4;        // 0..15
    const int tc4 = (threadIdx.x & 15) * 4;  // 0..60
    #pragma unroll
    for (int i = 0; i < 4; ++i) {
        int r = tr + i * 16;
        float4 v = *(const float4*)(src + (size_t)r * 1024 + tc4);
        tile[r * 72 + tc4 + 0] = (bf16_t)v.x;
        tile[r * 72 + tc4 + 1] = (bf16_t)v.y;
        tile[r * 72 + tc4 + 2] = (bf16_t)v.z;
        tile[r * 72 + tc4 + 3] = (bf16_t)v.w;
    }
    __syncthreads();
    bf16_t* dst = out + ((size_t)b * 1024 + s0) * 1024 + c0;
    #pragma unroll
    for (int j = 0; j < 2; ++j) {
        int chunk = threadIdx.x + j * 256;   // 512 chunks of 8
        int r  = chunk >> 3;                 // s-index 0..63
        int cc = (chunk & 7) * 8;            // c-chunk
        v8bf v;
        #pragma unroll
        for (int i = 0; i < 8; ++i) v[i] = tile[(cc + i) * 72 + r];
        *(v8bf*)(dst + (size_t)r * 1024 + cc) = v;
    }
}

// ---------------------------------------------------------------------------
// K0c: tiled bf16 transpose: h[b][t][s] -> hT[b][s][t] (64x64 tiles)
// ---------------------------------------------------------------------------
__global__ __launch_bounds__(256)
void transpose_bf16_kernel(const bf16_t* __restrict__ in, bf16_t* __restrict__ out)
{
    __shared__ __attribute__((aligned(16))) bf16_t tile[64 * 72];
    const int b  = blockIdx.z;
    const int t0 = blockIdx.y * 64;
    const int s0 = blockIdx.x * 64;
    const bf16_t* src = in + ((size_t)b * 1024 + t0) * 1024 + s0;
    #pragma unroll
    for (int j = 0; j < 2; ++j) {
        int chunk = threadIdx.x + j * 256;
        int r  = chunk >> 3;
        int cc = (chunk & 7) * 8;
        *(v8bf*)&tile[r * 72 + cc] = *(const v8bf*)(src + (size_t)r * 1024 + cc);
    }
    __syncthreads();
    bf16_t* dst = out + ((size_t)b * 1024 + s0) * 1024 + t0;
    #pragma unroll
    for (int j = 0; j < 2; ++j) {
        int chunk = threadIdx.x + j * 256;
        int r  = chunk >> 3;                 // s-index
        int cc = (chunk & 7) * 8;            // t-chunk
        v8bf v;
        #pragma unroll
        for (int i = 0; i < 8; ++i) v[i] = tile[(cc + i) * 72 + r];
        *(v8bf*)(dst + (size_t)r * 1024 + cc) = v;
    }
}

// ---------------------------------------------------------------------------
// K1: qkv = w_qkv @ x[b] (B operand = xT). Epilogue writes ALL outputs
// transposed (lane holds 8 contiguous o's) as single v8bf stores:
//   qT[b][s][o] (* scale^2), kT[b][s][o], vT[b][s][o]
// ---------------------------------------------------------------------------
__global__ __launch_bounds__(256)
void gemm_qkv_kernel(const bf16_t* __restrict__ wq, const bf16_t* __restrict__ xT,
                     bf16_t* __restrict__ qT, bf16_t* __restrict__ kT,
                     bf16_t* __restrict__ vT, float scale2)
{
    __shared__ __attribute__((aligned(16))) bf16_t As[2 * LDS_BUF];
    __shared__ __attribute__((aligned(16))) bf16_t Bs[2 * LDS_BUF];
    const int b  = blockIdx.z;
    const int m0 = blockIdx.y * BLK_M;     // o in [0,3072)
    const int n0 = blockIdx.x * BLK_N;     // s in [0,1024)
    v8f acc[4][2] = {};
    gemm_core_128x128(wq, 1024, xT + (size_t)b * 1024 * 1024, 1024, 1024,
                      m0, n0, As, Bs, acc);

    const int lane = threadIdx.x & 31;
    const int wave = threadIdx.x >> 5;
    const int wm = wave >> 2, wn = wave & 3;
    const int region = m0 >> 10;                    // 0=q,1=k,2=v (never straddles)
    bf16_t* dstT = (region == 0) ? qT : (region == 1) ? kT : vT;
    const float sc = (region == 0) ? scale2 : 1.0f;
    #pragma unroll
    for (int mi = 0; mi < 4; ++mi)
        #pragma unroll
        for (int ni = 0; ni < 2; ++ni) {
            int s     = n0 + wn * 32 + ni * 16 + (lane & 15);
            int obase = m0 - (region << 10) + wm * 64 + mi * 16 + (lane >> 4) * 8;
            v8bf pv;
            #pragma unroll
            for (int r = 0; r < 8; ++r) pv[r] = (bf16_t)(acc[mi][ni][r] * sc);
            *(v8bf*)&dstT[((size_t)b * 1024 + s) * 1024 + obase] = pv;
        }
}

// ---------------------------------------------------------------------------
// K2: vsum[b][s] = sum_c vT[b][s][c]   (wave-per-row, coalesced b128 reads)
// ---------------------------------------------------------------------------
__global__ __launch_bounds__(256)
void vsum_kernel(const bf16_t* __restrict__ vT, float* __restrict__ vsum)
{
    const int row  = blockIdx.x * 8 + (threadIdx.x >> 5);   // b*1024 + s
    const int lane = threadIdx.x & 31;
    const bf16_t* p = vT + (size_t)row * 1024;
    float a = 0.f;
    #pragma unroll
    for (int i = 0; i < 4; ++i) {
        v8bf v = *(const v8bf*)(p + (lane + i * 32) * 8);
        #pragma unroll
        for (int j = 0; j < 8; ++j) a += (float)v[j];
    }
    #pragma unroll
    for (int off = 16; off > 0; off >>= 1) a += __shfl_xor(a, off, 32);
    if (lane == 0) vsum[row] = a;
}

// ---------------------------------------------------------------------------
// K3: logits[b] = q^T k  == A=qT[t][c] row-major, BT=kT[s][c]
// ---------------------------------------------------------------------------
__global__ __launch_bounds__(256)
void gemm_logits_kernel(const bf16_t* __restrict__ qT, const bf16_t* __restrict__ kT,
                        float* __restrict__ logits)
{
    __shared__ __attribute__((aligned(16))) bf16_t As[2 * LDS_BUF];
    __shared__ __attribute__((aligned(16))) bf16_t Bs[2 * LDS_BUF];
    const int b  = blockIdx.z;
    const int m0 = blockIdx.y * BLK_M;     // t
    const int n0 = blockIdx.x * BLK_N;     // s
    v8f acc[4][2] = {};
    gemm_core_128x128(qT + (size_t)b * 1024 * 1024, 1024,
                      kT + (size_t)b * 1024 * 1024, 1024, 1024,
                      m0, n0, As, Bs, acc);
    const int lane = threadIdx.x & 31;
    const int wave = threadIdx.x >> 5;
    const int wm = wave >> 2, wn = wave & 3;
    #pragma unroll
    for (int mi = 0; mi < 4; ++mi)
        #pragma unroll
        for (int ni = 0; ni < 2; ++ni) {
            int s     = n0 + wn * 32 + ni * 16 + (lane & 15);
            int tbase = m0 + wm * 64 + mi * 16 + (lane >> 4) * 8;
            #pragma unroll
            for (int r = 0; r < 8; ++r)
                logits[((size_t)b * 1024 + tbase + r) * 1024 + s] = acc[mi][ni][r];
        }
}

// ---------------------------------------------------------------------------
// K4: h[b][t][s] = softmax_s(logits[b][t][:])[s] * vsum[b][s]  -> bf16
// ---------------------------------------------------------------------------
__global__ __launch_bounds__(256)
void softmax_vsum_kernel(const float* __restrict__ logits,
                         const float* __restrict__ vsum,
                         bf16_t* __restrict__ hb)
{
    __shared__ float red[256];
    const int row = blockIdx.x;            // b*1024 + t
    const int b   = row >> 10;
    const int t   = threadIdx.x;
    const float* lr = logits + (size_t)row * 1024;
    const float* vs = vsum + (size_t)b * 1024;

    float vals[4];
    float m = -3.0e38f;
    #pragma unroll
    for (int i = 0; i < 4; ++i) { vals[i] = lr[t + 256 * i]; m = fmaxf(m, vals[i]); }
    red[t] = m; __syncthreads();
    for (int sft = 128; sft > 0; sft >>= 1) {
        if (t < sft) red[t] = fmaxf(red[t], red[t + sft]);
        __syncthreads();
    }
    m = red[0]; __syncthreads();

    float sum = 0.f;
    #pragma unroll
    for (int i = 0; i < 4; ++i) { vals[i] = __expf(vals[i] - m); sum += vals[i]; }
    red[t] = sum; __syncthreads();
    for (int sft = 128; sft > 0; sft >>= 1) {
        if (t < sft) red[t] += red[t + sft];
        __syncthreads();
    }
    float inv = 1.0f / red[0];

    bf16_t* hr = hb + (size_t)row * 1024;
    #pragma unroll
    for (int i = 0; i < 4; ++i)
        hr[t + 256 * i] = (bf16_t)(vals[i] * inv * vs[t + 256 * i]);
}

// ---------------------------------------------------------------------------
// K5: out[b] = w_proj @ h[b] + b_proj + x[b]   (B operand = hT)
// ---------------------------------------------------------------------------
__global__ __launch_bounds__(256)
void gemm_out_kernel(const bf16_t* __restrict__ wp, const bf16_t* __restrict__ hT,
                     const float* __restrict__ bias, const float* __restrict__ x,
                     float* __restrict__ out)
{
    __shared__ __attribute__((aligned(16))) bf16_t As[2 * LDS_BUF];
    __shared__ __attribute__((aligned(16))) bf16_t Bs[2 * LDS_BUF];
    const int b  = blockIdx.z;
    const int m0 = blockIdx.y * BLK_M;     // o
    const int n0 = blockIdx.x * BLK_N;     // s
    v8f acc[4][2] = {};
    gemm_core_128x128(wp, 1024, hT + (size_t)b * 1024 * 1024, 1024, 1024,
                      m0, n0, As, Bs, acc);
    const int lane = threadIdx.x & 31;
    const int wave = threadIdx.x >> 5;
    const int wm = wave >> 2, wn = wave & 3;
    #pragma unroll
    for (int mi = 0; mi < 4; ++mi)
        #pragma unroll
        for (int ni = 0; ni < 2; ++ni) {
            int s     = n0 + wn * 32 + ni * 16 + (lane & 15);
            int obase = m0 + wm * 64 + mi * 16 + (lane >> 4) * 8;
            #pragma unroll
            for (int r = 0; r < 8; ++r) {
                int o = obase + r;
                size_t idx = ((size_t)b * 1024 + o) * 1024 + s;
                out[idx] = acc[mi][ni][r] + bias[o] + x[idx];
            }
        }
}

// ---------------------------------------------------------------------------
// Host launcher
// ---------------------------------------------------------------------------
extern "C" void kernel_launch(void* const* d_in, const int* in_sizes, int n_in,
                              void* d_out, int out_size, void* d_ws, size_t ws_size,
                              hipStream_t stream)
{
    (void)in_sizes; (void)n_in; (void)out_size; (void)ws_size;
    const float* x      = (const float*)d_in[0];   // [32,1024,32,32]
    const float* w_qkv  = (const float*)d_in[1];   // [3072,1024]
    const float* w_proj = (const float*)d_in[2];   // [1024,1024]
    const float* b_proj = (const float*)d_in[3];   // [1024]
    float* out = (float*)d_out;

    const size_t B = 32, C = 1024, S = 1024;
    char* ws = (char*)d_ws;
    size_t off = 0;
    bf16_t* wq_b = (bf16_t*)(ws + off); off += 3 * C * C * sizeof(bf16_t);   // 6 MB
    bf16_t* wp_b = (bf16_t*)(ws + off); off += C * C * sizeof(bf16_t);       // 2 MB
    bf16_t* qT   = (bf16_t*)(ws + off); off += B * S * C * sizeof(bf16_t);   // 64 MB
    bf16_t* kT   = (bf16_t*)(ws + off); off += B * S * C * sizeof(bf16_t);   // 64 MB
    float*  vsum = (float*)(ws + off);  off += B * S * sizeof(float);        // 128 KB
    bf16_t* xT   = (bf16_t*)(ws + off); off += B * S * C * sizeof(bf16_t);   // 64 MB
    bf16_t* vT   = (bf16_t*)(ws + off); off += B * S * C * sizeof(bf16_t);   // 64 MB
    // overlays (regions dead when these are written):
    float*  logits = (float*)xT;    // 128 MB over xT+vT (dead after GEMM1/vsum)
    bf16_t* hb     = qT;            // 64 MB over qT (dead after GEMM2)
    bf16_t* hT     = kT;            // 64 MB over kT (dead after GEMM2)

    {   // weights -> bf16 (vectorized: float4 -> 4x bf16)
        long long n4 = (long long)(3 * C * C) / 4;
        f32_to_bf16_kernel<<<(unsigned)((n4 + 255) / 256), 256, 0, stream>>>(w_qkv, wq_b, n4);
        n4 = (long long)(C * C) / 4;
        f32_to_bf16_kernel<<<(unsigned)((n4 + 255) / 256), 256, 0, stream>>>(w_proj, wp_b, n4);
    }
    // x -> xT bf16 (transposed, so GEMM1's B operand is K-major)
    transpose_convert_kernel<<<dim3(16, 16, 32), 256, 0, stream>>>(x, xT);

    const float scale2 = 0.03125f;  // hd^-0.5 = 1024^-0.5, folded into q once
    gemm_qkv_kernel<<<dim3(8, 24, 32), 256, 0, stream>>>(wq_b, xT, qT, kT, vT, scale2);
    vsum_kernel<<<(unsigned)(B * S / 8), 256, 0, stream>>>(vT, vsum);
    gemm_logits_kernel<<<dim3(8, 8, 32), 256, 0, stream>>>(qT, kT, logits);
    softmax_vsum_kernel<<<(unsigned)(B * S), 256, 0, stream>>>(logits, vsum, hb);
    transpose_bf16_kernel<<<dim3(16, 16, 32), 256, 0, stream>>>(hb, hT);
    gemm_out_kernel<<<dim3(8, 8, 32), 256, 0, stream>>>(wp_b, hT, b_proj, x, out);
}
